// BiLSTM_CRF_32959579029888
// MI455X (gfx1250) — compile-verified, matrix operand order checked by reading, not run on hardware
//
#include <hip/hip_runtime.h>
#include <hip/hip_bf16.h>

typedef _Float16 v16h __attribute__((ext_vector_type(16)));
typedef _Float16 v8h  __attribute__((ext_vector_type(8)));
typedef float    v8f  __attribute__((ext_vector_type(8)));

#define BB 64
#define TT 512
#define EE 768
#define HH 128
#define NG 512   // 4*H
#define KK 16

// ---------------------------------------------------------------------------
// helpers
// ---------------------------------------------------------------------------
__device__ inline v16h load16(const _Float16* p) {
    // lane holds K = [base..base+7] and [base+16..base+23] per CDNA5 16-bit
    // A/B operand layout (two b128 loads).
    v8h lo = *(const v8h*)p;
    v8h hi = *(const v8h*)(p + 16);
    return __builtin_shufflevector(lo, hi, 0,1,2,3,4,5,6,7,8,9,10,11,12,13,14,15);
}

__device__ inline v8f wmma16(v16h a, v16h b, v8f c) {
    return __builtin_amdgcn_wmma_f32_16x16x32_f16(
        /*neg_a=*/false, a, /*neg_b=*/false, b,
        /*c_mod=*/(short)0, c, /*reuse_a=*/false, /*reuse_b=*/false);
}

__device__ inline float sigf(float x) { return 1.0f / (1.0f + __expf(-x)); }

// ---------------------------------------------------------------------------
// prep kernels
// ---------------------------------------------------------------------------
__global__ void cvt_f16_kernel(const float* __restrict__ src,
                               _Float16* __restrict__ dst, int n) {
    int i = blockIdx.x * blockDim.x + threadIdx.x;
    if (i < n) dst[i] = (_Float16)src[i];
}

__global__ void add_bias_kernel(const float* __restrict__ a,
                                const float* __restrict__ b,
                                float* __restrict__ dst, int n) {
    int i = blockIdx.x * blockDim.x + threadIdx.x;
    if (i < n) dst[i] = a[i] + b[i];
}

// w_tag (256,16) f32 -> w_tagT (16,256) f16
__global__ void transpose_tag_kernel(const float* __restrict__ src,
                                     _Float16* __restrict__ dst) {
    int i = blockIdx.x * blockDim.x + threadIdx.x;
    if (i < KK * 256) {
        int n = i / 256, k = i % 256;
        dst[i] = (_Float16)src[k * KK + n];
    }
}

// x_f16[b,t,e] = emb[sentence[b,t], e] * mask[b,t]
__global__ void embed_kernel(const int* __restrict__ sentence,
                             const float* __restrict__ mask,
                             const float* __restrict__ emb,
                             _Float16* __restrict__ x, int total) {
    int i = blockIdx.x * blockDim.x + threadIdx.x;
    if (i >= total) return;
    int m = i / EE;
    int e = i - m * EE;
    int tok = sentence[m];
    x[i] = (_Float16)(emb[(size_t)tok * EE + e] * mask[m]);
}

// ---------------------------------------------------------------------------
// Y[m,n] = rowScale[m] * sum_k A[m,k]*W[n,k] + bias[n]
// A: M x Kd f16 row-major.  W: N x Kd f16 row-major (i.e. computes A @ W^T).
// Each wave computes a 16(M) x 16*NT(N) tile. If tmT>0, output row m=b*tmT+t
// is scattered to time-major index (t*Bv + b).
// ---------------------------------------------------------------------------
template<int NT>
__global__ __launch_bounds__(256) void wmma_gemm_kernel(
    const _Float16* __restrict__ A, const _Float16* __restrict__ W,
    const float* __restrict__ bias, const float* __restrict__ rowScale,
    float* __restrict__ Y, int M, int Kd, int N, int tmT, int Bv)
{
    const int lane   = threadIdx.x & 31;
    const int wid    = blockIdx.x * (blockDim.x >> 5) + (threadIdx.x >> 5);
    const int tilesN = N / (16 * NT);
    const int tm     = wid / tilesN;
    const int tn     = wid % tilesN;
    if (tm * 16 >= M) return;                 // wave-uniform, EXEC stays all-1s
    const int lr = lane & 15;
    const int hs = lane >> 4;                 // K-half select
    const int m0 = tm << 4;
    const int n0 = tn * (16 * NT);

    float bvs[NT];
#pragma unroll
    for (int nt = 0; nt < NT; ++nt)
        bvs[nt] = bias ? bias[n0 + nt * 16 + lr] : 0.0f;

    v8f acc[NT];
#pragma unroll
    for (int nt = 0; nt < NT; ++nt)
#pragma unroll
        for (int i = 0; i < 8; ++i) acc[nt][i] = 0.0f;

    const _Float16* Ap = A + (size_t)(m0 + lr) * Kd + (hs << 3);
    for (int k0 = 0; k0 < Kd; k0 += 32) {
        v16h a = load16(Ap + k0);
#pragma unroll
        for (int nt = 0; nt < NT; ++nt) {
            const _Float16* Wp = W + (size_t)(n0 + nt * 16 + lr) * Kd + (hs << 3);
            v16h b = load16(Wp + k0);
            acc[nt] = wmma16(a, b, acc[nt]);
        }
    }

#pragma unroll
    for (int i = 0; i < 8; ++i) {
        int   mo = m0 + (hs << 3) + i;        // output row per C/D layout
        float sr = rowScale ? rowScale[mo] : 1.0f;
        size_t base;
        if (tmT) { int t = mo % tmT; int bb = mo / tmT; base = ((size_t)t * Bv + bb) * N; }
        else      base = (size_t)mo * N;
#pragma unroll
        for (int nt = 0; nt < NT; ++nt)
            Y[base + n0 + nt * 16 + lr] = acc[nt][i] * sr + bvs[nt];
    }
}

// ---------------------------------------------------------------------------
// Persistent BiLSTM layer: blockIdx 0 = forward, 1 = reverse. One 1024-thread
// block per direction; per step gates = Gpre[t] + h @ Whh^T via WMMA with each
// wave holding the matching i/f/g/o tiles in registers. Whh lives in LDS
// (staged once, 128 KB — CDNA5 WGP has 320 KB), h in LDS (16 KB); B operands
// come from ds_load_b128 so nothing loop-invariant spills to scratch.
// Gpre: (T, 64, 512) f32 (bias folded in). Whh: (512, 128) f16.
// out:  (B, T, 256) f16, this direction at column offset dir*128.
// ---------------------------------------------------------------------------
__global__ __launch_bounds__(1024) void lstm_layer_kernel(
    const float* __restrict__ Gf, const float* __restrict__ Gr,
    const _Float16* __restrict__ Whf, const _Float16* __restrict__ Whr,
    _Float16* __restrict__ out)
{
    const int dir            = blockIdx.x;
    const float*    G        = dir ? Gr  : Gf;
    const _Float16* Whh      = dir ? Whr : Whf;
    const int       colOff   = dir * HH;

    __shared__ __align__(16) _Float16 hbuf[BB * HH];   // 16 KB
    __shared__ __align__(16) _Float16 wbuf[NG * HH];   // 128 KB (weights)

    const int tid  = threadIdx.x;
    const int wid  = tid >> 5;
    const int lane = tid & 31;
    const int tm   = wid >> 3;           // 0..3 -> batch tile
    const int tg   = wid & 7;            // 0..7 -> hidden tile
    const int m0   = tm << 4;
    const int n0   = tg << 4;
    const int lr   = lane & 15;
    const int hs   = lane >> 4;
    const int hi8  = hs << 3;

    // stage recurrent weights into LDS (b128 copies) + zero h
    for (int i = tid; i < (NG * HH) / 8; i += 1024)
        ((v8h*)wbuf)[i] = ((const v8h*)Whh)[i];
    for (int i = tid; i < BB * HH; i += 1024) hbuf[i] = (_Float16)0.0f;
    __syncthreads();

    float c[8];
#pragma unroll
    for (int i = 0; i < 8; ++i) c[i] = 0.0f;

    for (int s = 0; s < TT; ++s) {
        const int t = dir ? (TT - 1 - s) : s;
        const float* Gt = G + (size_t)t * BB * NG;

        // prefetch next step's 128KB Gpre slab while this step computes
        if (s + 1 < TT) {
            const int tn_ = dir ? (TT - 2 - s) : (s + 1);
            const float* Gn = G + (size_t)tn_ * BB * NG;
#pragma unroll
            for (int q = 0; q < 4; ++q)
                __builtin_prefetch(Gn + ((size_t)tid * 4 + q) * 32, 0, 3);
        }

        v8f acc[4];                       // i, f, g, o tiles (same 16 hidden)
#pragma unroll
        for (int g = 0; g < 4; ++g) {
            const int col = g * HH + n0 + lr;
#pragma unroll
            for (int i = 0; i < 8; ++i)
                acc[g][i] = Gt[(m0 + hi8 + i) * NG + col];
        }

#pragma unroll
        for (int k0 = 0; k0 < HH; k0 += 32) {
            v16h a = load16(&hbuf[(m0 + lr) * HH + hi8 + k0]);
#pragma unroll
            for (int g = 0; g < 4; ++g) {
                const _Float16* Wp = &wbuf[(g * HH + n0 + lr) * HH + hi8];
                v16h b = load16(Wp + k0);
                acc[g] = wmma16(a, b, acc[g]);
            }
        }

        __syncthreads();                  // all h reads done before overwrite
#pragma unroll
        for (int i = 0; i < 8; ++i) {
            const int row = m0 + hi8 + i;
            const int col = n0 + lr;
            float gi = sigf(acc[0][i]);
            float gf = sigf(acc[1][i]);
            float gg = tanhf(acc[2][i]);
            float go = sigf(acc[3][i]);
            c[i] = gf * c[i] + gi * gg;
            float h = go * tanhf(c[i]);
            _Float16 hh = (_Float16)h;
            hbuf[row * HH + col] = hh;
            out[((size_t)row * TT + t) * (2 * HH) + colOff + col] = hh;
        }
        __syncthreads();
    }
}

// ---------------------------------------------------------------------------
// CRF: one wave per batch row. Lanes 0..15 hold alpha; shfl-based logsumexp.
// ---------------------------------------------------------------------------
__global__ void crf_kernel(const float* __restrict__ em,
                           const int* __restrict__ tags,
                           const float* __restrict__ mask,
                           const float* __restrict__ trans,
                           const float* __restrict__ start_t,
                           const float* __restrict__ end_t,
                           float* __restrict__ perb)
{
    const int b  = blockIdx.x;
    const int j  = threadIdx.x;          // 0..31
    const int jc = j & 15;
    const float* emB = em + (size_t)b * TT * KK;
    const int*   tg  = tags + b * TT;
    const float* mk  = mask + b * TT;

    float alpha = (j < 16) ? (start_t[jc] + emB[jc]) : -1e30f;

    for (int t = 1; t < TT; ++t) {
        float mt = mk[t];
        float best = -1e30f;
#pragma unroll 4
        for (int i = 0; i < 16; ++i) {
            float ai = __shfl(alpha, i, 32);
            best = fmaxf(best, ai + trans[i * KK + jc]);
        }
        float ss = 0.0f;
#pragma unroll 4
        for (int i = 0; i < 16; ++i) {
            float ai = __shfl(alpha, i, 32);
            ss += __expf(ai + trans[i * KK + jc] - best);
        }
        float nxt = best + __logf(ss) + emB[t * KK + jc];
        if (mt > 0.0f) alpha = nxt;      // mt uniform across the wave
    }

    float v  = (j < 16) ? (alpha + end_t[jc]) : -1e30f;
    float mx = v;
#pragma unroll
    for (int off = 16; off >= 1; off >>= 1) mx = fmaxf(mx, __shfl_xor(mx, off, 32));
    float se = __expf(v - mx);
#pragma unroll
    for (int off = 16; off >= 1; off >>= 1) se += __shfl_xor(se, off, 32);
    float logZ = mx + __logf(se);

    if (j == 0) {
        float num = start_t[tg[0]] + emB[tg[0]];
        int sl = 0;
        for (int t = 0; t < TT; ++t) sl += (mk[t] > 0.0f) ? 1 : 0;
        for (int t = 1; t < TT; ++t)
            num += (emB[t * KK + tg[t]] + trans[tg[t - 1] * KK + tg[t]]) * mk[t];
        num += end_t[tg[sl - 1]];
        perb[b] = logZ - num;
    }
}

__global__ void reduce_mean_kernel(const float* __restrict__ perb,
                                   float* __restrict__ out, int n) {
    if (blockIdx.x == 0 && threadIdx.x == 0) {
        float s = 0.0f;
        for (int i = 0; i < n; ++i) s += perb[i];   // fixed order: deterministic
        out[0] = s / (float)n;
    }
}

// ---------------------------------------------------------------------------
// host launcher
// ---------------------------------------------------------------------------
extern "C" void kernel_launch(void* const* d_in, const int* in_sizes, int n_in,
                              void* d_out, int out_size, void* d_ws, size_t ws_size,
                              hipStream_t stream) {
    (void)in_sizes; (void)n_in; (void)out_size; (void)ws_size;

    const int*   sentence = (const int*)  d_in[0];
    const int*   tags     = (const int*)  d_in[1];
    const float* mask     = (const float*)d_in[2];
    const float* emb      = (const float*)d_in[3];
    const float* w_ih_l0  = (const float*)d_in[4];
    const float* w_hh_l0  = (const float*)d_in[5];
    const float* b_ih_l0  = (const float*)d_in[6];
    const float* b_hh_l0  = (const float*)d_in[7];
    const float* w_ih_l0r = (const float*)d_in[8];
    const float* w_hh_l0r = (const float*)d_in[9];
    const float* b_ih_l0r = (const float*)d_in[10];
    const float* b_hh_l0r = (const float*)d_in[11];
    const float* w_ih_l1  = (const float*)d_in[12];
    const float* w_hh_l1  = (const float*)d_in[13];
    const float* b_ih_l1  = (const float*)d_in[14];
    const float* b_hh_l1  = (const float*)d_in[15];
    const float* w_ih_l1r = (const float*)d_in[16];
    const float* w_hh_l1r = (const float*)d_in[17];
    const float* b_ih_l1r = (const float*)d_in[18];
    const float* b_hh_l1r = (const float*)d_in[19];
    const float* w_tag    = (const float*)d_in[20];
    const float* b_tag    = (const float*)d_in[21];
    const float* trans    = (const float*)d_in[22];
    const float* start_t  = (const float*)d_in[23];
    const float* end_t    = (const float*)d_in[24];
    float* out = (float*)d_out;

    // workspace carve-up (256B aligned)
    char* wp = (char*)d_ws;
    auto carve = [&](size_t bytes) -> void* {
        void* r = (void*)wp;
        wp += (bytes + 255) & ~(size_t)255;
        return r;
    };
    const size_t M = (size_t)BB * TT;               // 32768 rows
    _Float16* x16    = (_Float16*)carve(M * EE  * 2);
    _Float16* x1     = (_Float16*)carve(M * 256 * 2);
    _Float16* enc    = (_Float16*)carve(M * 256 * 2);
    float*    Gf     = (float*)   carve((size_t)TT * BB * NG * 4);
    float*    Gr     = (float*)   carve((size_t)TT * BB * NG * 4);
    float*    emis   = (float*)   carve(M * KK * 4);
    _Float16* w0f16  = (_Float16*)carve((size_t)NG * EE  * 2);
    _Float16* w0r16  = (_Float16*)carve((size_t)NG * EE  * 2);
    _Float16* w1f16  = (_Float16*)carve((size_t)NG * 256 * 2);
    _Float16* w1r16  = (_Float16*)carve((size_t)NG * 256 * 2);
    _Float16* wh0f   = (_Float16*)carve((size_t)NG * HH * 2);
    _Float16* wh0r   = (_Float16*)carve((size_t)NG * HH * 2);
    _Float16* wh1f   = (_Float16*)carve((size_t)NG * HH * 2);
    _Float16* wh1r   = (_Float16*)carve((size_t)NG * HH * 2);
    _Float16* wtagT  = (_Float16*)carve((size_t)KK * 256 * 2);
    float*    bias0f = (float*)carve(NG * 4);
    float*    bias0r = (float*)carve(NG * 4);
    float*    bias1f = (float*)carve(NG * 4);
    float*    bias1r = (float*)carve(NG * 4);
    float*    perb   = (float*)carve(BB * 4);

    auto blocksFor = [](int n) { return (n + 255) / 256; };

    // ---- weight prep -------------------------------------------------------
    cvt_f16_kernel<<<blocksFor(NG*EE ), 256, 0, stream>>>(w_ih_l0 , w0f16, NG*EE );
    cvt_f16_kernel<<<blocksFor(NG*EE ), 256, 0, stream>>>(w_ih_l0r, w0r16, NG*EE );
    cvt_f16_kernel<<<blocksFor(NG*256), 256, 0, stream>>>(w_ih_l1 , w1f16, NG*256);
    cvt_f16_kernel<<<blocksFor(NG*256), 256, 0, stream>>>(w_ih_l1r, w1r16, NG*256);
    cvt_f16_kernel<<<blocksFor(NG*HH ), 256, 0, stream>>>(w_hh_l0 , wh0f , NG*HH );
    cvt_f16_kernel<<<blocksFor(NG*HH ), 256, 0, stream>>>(w_hh_l0r, wh0r , NG*HH );
    cvt_f16_kernel<<<blocksFor(NG*HH ), 256, 0, stream>>>(w_hh_l1 , wh1f , NG*HH );
    cvt_f16_kernel<<<blocksFor(NG*HH ), 256, 0, stream>>>(w_hh_l1r, wh1r , NG*HH );
    add_bias_kernel<<<2, 256, 0, stream>>>(b_ih_l0 , b_hh_l0 , bias0f, NG);
    add_bias_kernel<<<2, 256, 0, stream>>>(b_ih_l0r, b_hh_l0r, bias0r, NG);
    add_bias_kernel<<<2, 256, 0, stream>>>(b_ih_l1 , b_hh_l1 , bias1f, NG);
    add_bias_kernel<<<2, 256, 0, stream>>>(b_ih_l1r, b_hh_l1r, bias1r, NG);
    transpose_tag_kernel<<<blocksFor(KK*256), 256, 0, stream>>>(w_tag, wtagT);

    // ---- embedding gather (+mask) -----------------------------------------
    {
        int total = (int)(M * EE);
        embed_kernel<<<blocksFor(total), 256, 0, stream>>>(sentence, mask, emb, x16, total);
    }

    // ---- layer 0 input projections: (T,64,512) time-major -----------------
    {
        int waves  = ((int)M / 16) * (NG / 64);
        int blocks = waves / 8;
        wmma_gemm_kernel<4><<<blocks, 256, 0, stream>>>(x16, w0f16, bias0f, nullptr,
                                                        Gf, (int)M, EE, NG, TT, BB);
        wmma_gemm_kernel<4><<<blocks, 256, 0, stream>>>(x16, w0r16, bias0r, nullptr,
                                                        Gr, (int)M, EE, NG, TT, BB);
    }
    lstm_layer_kernel<<<2, 1024, 0, stream>>>(Gf, Gr, wh0f, wh0r, x1);

    // ---- layer 1 input projections ----------------------------------------
    {
        int waves  = ((int)M / 16) * (NG / 64);
        int blocks = waves / 8;
        wmma_gemm_kernel<4><<<blocks, 256, 0, stream>>>(x1, w1f16, bias1f, nullptr,
                                                        Gf, (int)M, 256, NG, TT, BB);
        wmma_gemm_kernel<4><<<blocks, 256, 0, stream>>>(x1, w1r16, bias1r, nullptr,
                                                        Gr, (int)M, 256, NG, TT, BB);
    }
    lstm_layer_kernel<<<2, 1024, 0, stream>>>(Gf, Gr, wh1f, wh1r, enc);

    // ---- emissions = (enc * mask) @ w_tag + b_tag -------------------------
    {
        int waves  = ((int)M / 16);            // tilesN == 1 for NT=1, N=16
        int blocks = waves / 8;
        wmma_gemm_kernel<1><<<blocks, 256, 0, stream>>>(enc, wtagT, b_tag, mask,
                                                        emis, (int)M, 256, KK, 0, 0);
    }

    // ---- CRF NLL -----------------------------------------------------------
    crf_kernel<<<BB, 32, 0, stream>>>(emis, tags, mask, trans, start_t, end_t, perb);
    reduce_mean_kernel<<<1, 64, 0, stream>>>(perb, out, BB);
}
